// LSTMModel_7816840479375
// MI455X (gfx1250) — matrix operand order repaired and checked by reading, not compiled, not run on hardware
//
#include <hip/hip_runtime.h>

// ---------------------------------------------------------------------------
// Fused 2-layer LSTM + FC head for MI455X (gfx1250, wave32, WMMA).
//   B=512, T=512, IN=32, H=64, gates G=4H=256.
// One workgroup per 16 batch rows; 16 waves, each owning a 16-column gate
// tile. Weights persistent in VGPRs as WMMA B-fragments (f16), h state in
// LDS, c state in registers. Bias folded into WMMA accumulator init.
// x streamed with async global->LDS (double-buffered 8-step chunks,
// global_load_async_to_lds_b128 + s_wait_asynccnt). Branch-free activations
// (v_tanh_f32 when available).
// ---------------------------------------------------------------------------

typedef __attribute__((ext_vector_type(16))) _Float16 v16h;
typedef __attribute__((ext_vector_type(8)))  _Float16 v8h;
typedef __attribute__((ext_vector_type(8)))  float    v8f;
typedef __attribute__((ext_vector_type(4)))  float    v4f;

#define WMMA_F16(a, b, c) \
    __builtin_amdgcn_wmma_f32_16x16x32_f16(false, (a), false, (b), (short)0, (c), false, false)

constexpr int B_  = 512;
constexpr int T_  = 512;
constexpr int IN_ = 32;
constexpr int H_  = 64;
constexpr int G_  = 4 * H_;   // 256 gate columns
constexpr int CHUNK = 8;      // timesteps of x staged per async refill

// ---- branch-free activations (1 TRANS op each when v_tanh_f32 exists) -----
#if __has_builtin(__builtin_amdgcn_tanhf)
__device__ __forceinline__ float fast_tanh(float x) { return __builtin_amdgcn_tanhf(x); }
#else
__device__ __forceinline__ float fast_tanh(float x) {
    // tanh(x) = 1 - 2/(exp(2x)+1), branchless: exp + rcp
    return 1.0f - 2.0f * __builtin_amdgcn_rcpf(__expf(2.0f * x) + 1.0f);
}
#endif
__device__ __forceinline__ float fast_sigm(float x) { return 0.5f + 0.5f * fast_tanh(0.5f * x); }

// ---- async global -> LDS (CDNA5), GV addressing mode -----------------------
__device__ __forceinline__ void async_ld_b128(unsigned int lds_addr, unsigned long long gaddr) {
    asm volatile("global_load_async_to_lds_b128 %0, %1, off"
                 :: "v"(lds_addr), "v"(gaddr) : "memory");
}
__device__ __forceinline__ void wait_async_le2() {
#if __has_builtin(__builtin_amdgcn_s_wait_asynccnt)
    __builtin_amdgcn_s_wait_asynccnt(2);
#else
    asm volatile("s_wait_asynccnt 0x2" ::: "memory");
#endif
}
__device__ __forceinline__ void wait_async_0() {
#if __has_builtin(__builtin_amdgcn_s_wait_asynccnt)
    __builtin_amdgcn_s_wait_asynccnt(0);
#else
    asm volatile("s_wait_asynccnt 0x0" ::: "memory");
#endif
}

// A-fragment (16x32 f16) from f16 LDS. ISA 7.12.2 16-bit A layout:
//   lanes 0-15 : row = lane,    K = kb + {0..7, 16..23}
//   lanes 16-31: row = lane-16, K = kb + {8..15, 24..31}
__device__ __forceinline__ v16h load_a16(const _Float16* base, int stride, int lane, int kb) {
    const int row  = lane & 15;
    const int koff = (lane >> 4) << 3;                 // 0 or 8
    const _Float16* p = base + row * stride + kb + koff;
    v8h lo = *reinterpret_cast<const v8h*>(p);
    v8h hi = *reinterpret_cast<const v8h*>(p + 16);
    v16h a;
#pragma unroll
    for (int i = 0; i < 8; i++) { a[i] = lo[i]; a[i + 8] = hi[i]; }
    return a;
}

// A-fragment from f32 LDS (x tile), converting to f16 at load.
__device__ __forceinline__ v16h load_a16_f32(const float* base, int stride, int lane) {
    const int row  = lane & 15;
    const int koff = (lane >> 4) << 3;                 // 0 or 8
    const float* p = base + row * stride + koff;
    v4f a0 = *reinterpret_cast<const v4f*>(p);
    v4f a1 = *reinterpret_cast<const v4f*>(p + 4);
    v4f b0 = *reinterpret_cast<const v4f*>(p + 16);
    v4f b1 = *reinterpret_cast<const v4f*>(p + 20);
    v16h a;
#pragma unroll
    for (int i = 0; i < 4; i++) {
        a[i]      = (_Float16)a0[i];
        a[i + 4]  = (_Float16)a1[i];
        a[i + 8]  = (_Float16)b0[i];
        a[i + 12] = (_Float16)b1[i];
    }
    return a;
}

// B-fragment from one f32 weight row W[n][*]: K = kstart + {0..7, 16..23}.
__device__ __forceinline__ v16h load_w16(const float* __restrict__ wrow, int kstart) {
    v16h f;
#pragma unroll
    for (int i = 0; i < 16; i++) f[i] = (_Float16)wrow[kstart + (i < 8 ? i : i + 8)];
    return f;
}

__global__ __launch_bounds__(512, 1)
void lstm2_fused_kernel(const float* __restrict__ x,
                        const float* __restrict__ Wih0, const float* __restrict__ Whh0,
                        const float* __restrict__ bih0, const float* __restrict__ bhh0,
                        const float* __restrict__ Wih1, const float* __restrict__ Whh1,
                        const float* __restrict__ bih1, const float* __restrict__ bhh1,
                        const float* __restrict__ Wfc,  const float* __restrict__ bfc,
                        float* __restrict__ out)
{
    // x staged as f32 via async DMA: [buf][row][step][col]; row segment = 1KB contiguous
    __shared__ __align__(16) float    xsf[2][16][CHUNK][IN_];
    __shared__ __align__(16) _Float16 h0s[16][H_];
    __shared__ __align__(16) _Float16 h1s[16][H_];
    __shared__ float g0s[16][G_];
    __shared__ float g1s[16][G_];
    __shared__ float outp[16];

    const int tid  = threadIdx.x;          // 0..511
    const int lane = tid & 31;
    const int wid  = tid >> 5;             // 0..15 : gate-column tile owner
    const int n0   = wid * 16;             // first gate column of this wave
    const int r0   = blockIdx.x * 16;      // first batch row of this WG

    // ---- one-time init -----------------------------------------------------
    if (tid < 16) outp[tid] = 0.0f;
    {
        _Float16* z0 = &h0s[0][0];
        _Float16* z1 = &h1s[0][0];
        z0[tid] = (_Float16)0.f; z0[tid + 512] = (_Float16)0.f;
        z1[tid] = (_Float16)0.f; z1[tid + 512] = (_Float16)0.f;
    }

    // ---- persistent WMMA B-fragments (weights, f32 -> f16, loaded once) ----
    const int col  = n0 + (lane & 15);     // gate column this lane owns in B/C/D
    const int kh   = (lane >> 4) * 8;      // K offset (0 or 8) of this lane half
    v16h fWih0 = load_w16(Wih0 + col * IN_, kh);           // K=32 (input GEMM L0)
    v16h fWhh0[2], fWih1[2], fWhh1[2];                     // K=64 -> 2 K-blocks
#pragma unroll
    for (int b = 0; b < 2; b++) {
        fWhh0[b] = load_w16(Whh0 + col * H_, b * 32 + kh);
        fWih1[b] = load_w16(Wih1 + col * H_, b * 32 + kh);
        fWhh1[b] = load_w16(Whh1 + col * H_, b * 32 + kh);
    }
    // bias folded into WMMA accumulator init (per-lane, column-only dependence)
    const float bcol0 = bih0[col] + bhh0[col];
    const float bcol1 = bih1[col] + bhh1[col];

    // elementwise ownership: thread handles cells (er, ej) and (er+8, ej)
    const int ej = tid & 63;               // hidden unit
    const int er = tid >> 6;               // 0..7 batch row within tile
    const float wfcj = Wfc[ej];
    float c0[2] = {0.f, 0.f};
    float c1[2] = {0.f, 0.f};

    // ---- async staging: each thread DMAs 2x16B of its wave's batch row -----
    const int xrow = wid;                                   // row this wave stages
    const unsigned int seg = (unsigned int)(lane) * 16u;    // 16B piece within 1KB row seg
    const float* xrow_base = x + (size_t)(r0 + xrow) * T_ * IN_;

    auto issue_chunk = [&](int buf, int t) {
        unsigned long long g =
            (unsigned long long)(const void*)(xrow_base + (size_t)t * IN_);
        unsigned int l = (unsigned int)(unsigned long long)(const void*)&xsf[buf][xrow][0][0];
        async_ld_b128(l + seg,        g + seg);
        async_ld_b128(l + seg + 512u, g + seg + 512u);
    };

    issue_chunk(0, 0);                     // prefetch chunk 0
    __syncthreads();

    // ---- recurrence over T -------------------------------------------------
    for (int t = 0; t < T_; t++) {
        const int tc = t & (CHUNK - 1);
        const int cc = t >> 3;             // chunk index
        if (tc == 0) {
            if (t + CHUNK < T_) {          // prefetch next chunk into other buffer
                issue_chunk((cc + 1) & 1, t + CHUNK);
                wait_async_le2();          // current chunk's 2 asyncs complete (in-order)
            } else {
                wait_async_0();
            }
            __syncthreads();               // make async-written LDS visible to all waves
        }
        const float* xtile = &xsf[cc & 1][0][tc][0];   // row stride = CHUNK*IN_ floats

        // ---- layer 0: gates = bias + x_t @ Wih0^T + h0 @ Whh0^T ----
        v8f acc = {bcol0, bcol0, bcol0, bcol0, bcol0, bcol0, bcol0, bcol0};
        acc = WMMA_F16(load_a16_f32(xtile, CHUNK * IN_, lane), fWih0,    acc);
        acc = WMMA_F16(load_a16(&h0s[0][0], H_, lane, 0),      fWhh0[0], acc);
        acc = WMMA_F16(load_a16(&h0s[0][0], H_, lane, 32),     fWhh0[1], acc);
        {
            const int rb = (lane < 16) ? 0 : 8;    // C/D layout: VGPR v -> M = v (+8)
#pragma unroll
            for (int v = 0; v < 8; v++) g0s[rb + v][col] = acc[v];
        }
        __syncthreads();

        // ---- layer 0 elementwise: i,f,g,o -> c0, h0 ----
#pragma unroll
        for (int q = 0; q < 2; q++) {
            const int row = er + q * 8;
            const float i_ = fast_sigm(g0s[row][          ej]);
            const float f_ = fast_sigm(g0s[row][    H_ + ej]);
            const float g_ = fast_tanh(g0s[row][2 * H_ + ej]);
            const float o_ = fast_sigm(g0s[row][3 * H_ + ej]);
            const float cv = f_ * c0[q] + i_ * g_;
            c0[q] = cv;
            h0s[row][ej] = (_Float16)(o_ * fast_tanh(cv));
        }
        __syncthreads();

        // ---- layer 1: gates = bias + h0_t @ Wih1^T + h1 @ Whh1^T ----
        v8f acc1 = {bcol1, bcol1, bcol1, bcol1, bcol1, bcol1, bcol1, bcol1};
        acc1 = WMMA_F16(load_a16(&h0s[0][0], H_, lane, 0),  fWih1[0], acc1);
        acc1 = WMMA_F16(load_a16(&h0s[0][0], H_, lane, 32), fWih1[1], acc1);
        acc1 = WMMA_F16(load_a16(&h1s[0][0], H_, lane, 0),  fWhh1[0], acc1);
        acc1 = WMMA_F16(load_a16(&h1s[0][0], H_, lane, 32), fWhh1[1], acc1);
        {
            const int rb = (lane < 16) ? 0 : 8;
#pragma unroll
            for (int v = 0; v < 8; v++) g1s[rb + v][col] = acc1[v];
        }
        __syncthreads();

        // ---- layer 1 elementwise: -> c1, h1 (+ FC head at t = T-1) ----
#pragma unroll
        for (int q = 0; q < 2; q++) {
            const int row = er + q * 8;
            const float i_ = fast_sigm(g1s[row][          ej]);
            const float f_ = fast_sigm(g1s[row][    H_ + ej]);
            const float g_ = fast_tanh(g1s[row][2 * H_ + ej]);
            const float o_ = fast_sigm(g1s[row][3 * H_ + ej]);
            const float cv = f_ * c1[q] + i_ * g_;
            c1[q] = cv;
            const float hv = o_ * fast_tanh(cv);
            h1s[row][ej] = (_Float16)hv;
            if (t == T_ - 1) atomicAdd(&outp[row], hv * wfcj);  // ds_add_f32
        }
        __syncthreads();
    }

    if (tid < 16) out[r0 + tid] = outp[tid] + bfc[0];
}

extern "C" void kernel_launch(void* const* d_in, const int* in_sizes, int n_in,
                              void* d_out, int out_size, void* d_ws, size_t ws_size,
                              hipStream_t stream) {
    (void)in_sizes; (void)n_in; (void)out_size; (void)d_ws; (void)ws_size;
    const float* x    = (const float*)d_in[0];
    const float* Wih0 = (const float*)d_in[1];
    const float* Whh0 = (const float*)d_in[2];
    const float* bih0 = (const float*)d_in[3];
    const float* bhh0 = (const float*)d_in[4];
    const float* Wih1 = (const float*)d_in[5];
    const float* Whh1 = (const float*)d_in[6];
    const float* bih1 = (const float*)d_in[7];
    const float* bhh1 = (const float*)d_in[8];
    const float* Wfc  = (const float*)d_in[9];
    const float* bfc  = (const float*)d_in[10];
    float* out = (float*)d_out;

    lstm2_fused_kernel<<<dim3(B_ / 16), dim3(512), 0, stream>>>(
        x, Wih0, Whh0, bih0, bhh0, Wih1, Whh1, bih1, bhh1, Wfc, bfc, out);
}
